// SpaGCN_Transformer_79422535237686
// MI455X (gfx1250) — matrix operand order repaired and checked by reading, not compiled, not run on hardware
//
#include <hip/hip_runtime.h>
#include <hip/hip_bf16.h>
#include <math.h>

typedef __attribute__((ext_vector_type(16))) __bf16 v16bf;
typedef __attribute__((ext_vector_type(8)))  float  v8f;

#define BN_EPS 1e-5f
#define CIN 512
#define GH 128
#define GO 64
#define TH 64
#define HEADS 4
#define COUT 32

// ---------------------------------------------------------------- utilities
__global__ void fill_kernel(float* __restrict__ p, float v, long n) {
  long i = (long)blockIdx.x * blockDim.x + threadIdx.x;
  if (i < n) p[i] = v;
}

__global__ void deg_accum_kernel(const int* __restrict__ dst,
                                 const float* __restrict__ w,
                                 float* __restrict__ deg, int E) {
  int e = blockIdx.x * blockDim.x + threadIdx.x;
  if (e < E) atomicAdd(&deg[dst[e]], w[e]);
}

__global__ void dinv_kernel(const float* __restrict__ deg,
                            float* __restrict__ dinv, int n) {
  int i = blockIdx.x * blockDim.x + threadIdx.x;
  if (i < n) {
    float d = deg[i];
    dinv[i] = (d > 0.f) ? rsqrtf(d) : 0.f;
  }
}

// ---------------------------------------------------------------- B pre-swizzle
// Pack fp32 weights B[K,Nc] into bf16 WMMA-fragment order:
//   Bp[ (ntile*(K/32)+kt)*32 + lane ][ j ]  with
//   lane = 16*half + l16 -> element K = kt*32 + (j<8 ? 8*half+j : 16+8*half+j-8), col = ntile*16+l16.
// Each lane then loads its 16 bf16 (32 B, contiguous) with two b128 loads.
template <int K, int Nc>
__global__ __launch_bounds__(512)
void pack_B_bf16(const float* __restrict__ B, __bf16* __restrict__ Bp) {
  const int tile = blockIdx.x;          // ntile*(K/32) + kt
  const int ntile = tile / (K / 32);
  const int kt = tile % (K / 32);
  const int lane = threadIdx.y;         // 0..31
  const int j = threadIdx.x;            // 0..15
  const int half = lane >> 4, l16 = lane & 15;
  const int kk = kt * 32 + (j < 8 ? 8 * half + j : 16 + 8 * half + (j - 8));
  const int col = ntile * 16 + l16;
  Bp[((size_t)tile * 32 + lane) * 16 + j] = (__bf16)B[(size_t)kk * Nc + col];
}

// ---------------------------------------------------------------- WMMA GEMM
// C[M,Nc] = A[M,K] @ B[K,Nc] (+bias), fp32 A/C, pre-swizzled bf16 B.
// One wave = one 16x16 tile; 8 waves/block reuse the same A slab from WGP$/L2.
// Inner loop: 4x b128 A loads + 2x b128 B loads + 8 pk-cvt + 1 WMMA.
template <int K, int Nc>
__global__ __launch_bounds__(256)
void gemm_bf16_wmma(const float* __restrict__ A, const __bf16* __restrict__ Bp,
                    const float* __restrict__ bias, float* __restrict__ C,
                    int M) {
  const int lane = threadIdx.x & 31;
  const int half = lane >> 4;       // 0: lanes 0-15, 1: lanes 16-31
  const int l16  = lane & 15;
  const int ntile = blockIdx.y * 8 + threadIdx.y;
  if (ntile * 16 >= Nc) return;     // uniform per wave (threadIdx.y uniform)
  const int m0 = blockIdx.x * 16;
  const int n0 = ntile * 16;
  int rowA = m0 + l16;
  if (rowA >= M) rowA = M - 1;      // clamp (branch-free); stores are guarded
  const int colB = n0 + l16;
  const float* __restrict__ Arow = A + (size_t)rowA * K;
  const __bf16* __restrict__ BpLane =
      Bp + ((size_t)ntile * (K / 32) * 32 + lane) * 16;

  v8f acc = {};
#pragma unroll 2
  for (int kt = 0; kt < K / 32; ++kt) {
    const int k0 = kt * 32;
    if (k0 + 64 <= K)
      __builtin_prefetch(&Arow[k0 + 32], 0, 1);
    // A fragment: lane holds K in [k0+8*half, +8) and [k0+16+8*half, +8),
    // both contiguous -> four b128 loads with immediate offsets.
    const float4 a0 = *(const float4*)(Arow + k0 + 8 * half);
    const float4 a1 = *(const float4*)(Arow + k0 + 8 * half + 4);
    const float4 a2 = *(const float4*)(Arow + k0 + 16 + 8 * half);
    const float4 a3 = *(const float4*)(Arow + k0 + 16 + 8 * half + 4);
    v16bf af;
    af[0]  = (__bf16)a0.x; af[1]  = (__bf16)a0.y; af[2]  = (__bf16)a0.z; af[3]  = (__bf16)a0.w;
    af[4]  = (__bf16)a1.x; af[5]  = (__bf16)a1.y; af[6]  = (__bf16)a1.z; af[7]  = (__bf16)a1.w;
    af[8]  = (__bf16)a2.x; af[9]  = (__bf16)a2.y; af[10] = (__bf16)a2.z; af[11] = (__bf16)a2.w;
    af[12] = (__bf16)a3.x; af[13] = (__bf16)a3.y; af[14] = (__bf16)a3.z; af[15] = (__bf16)a3.w;
    // B fragment: 16 bf16 contiguous per lane (32 B, 32B-aligned) -> 2x b128.
    const v16bf bfv = *(const v16bf*)(BpLane + (size_t)kt * 32 * 16);
    acc = __builtin_amdgcn_wmma_f32_16x16x32_bf16(
        /*neg_a=*/false, af, /*neg_b=*/false, bfv,
        /*c_mod=*/(short)0, acc, /*reuse_a=*/false, /*reuse_b=*/false);
  }
  const float bv = bias ? bias[colB] : 0.f;
  float* __restrict__ Cp = C + (size_t)(m0 + 8 * half) * Nc + colB;
#pragma unroll
  for (int v = 0; v < 8; ++v) {     // C/D layout: VGPR v -> row v + 8*half
    if (m0 + v + 8 * half < M) Cp[v * Nc] = acc[v] + bv;
  }
}

// ---------------------------------------------------------------- GCN pieces
__global__ void gcn_edge_agg(const int* __restrict__ src, const int* __restrict__ dst,
                             const float* __restrict__ w, const float* __restrict__ dinv,
                             const float* __restrict__ h, float* __restrict__ out, int F) {
  int e = blockIdx.x;
  int f = threadIdx.x;
  int s = src[e], d = dst[e];
  float nrm = dinv[s] * w[e] * dinv[d];
  atomicAdd(&out[(long)d * F + f], nrm * h[(long)s * F + f]);
}

// self-loop + bias + BN(eval) + ReLU ; out may alias agg (elementwise)
__global__ void gcn_fuse(const float* __restrict__ agg, const float* __restrict__ h,
                         const float* __restrict__ dinv, const float* __restrict__ b,
                         const float* __restrict__ g, const float* __restrict__ be,
                         const float* __restrict__ m, const float* __restrict__ v,
                         float* __restrict__ out, int F) {
  int i = blockIdx.x;
  int f = threadIdx.x;
  float self = dinv[i] * dinv[i] * h[(long)i * F + f];
  float val = agg[(long)i * F + f] + self + b[f];
  float s = g[f] * rsqrtf(v[f] + BN_EPS);
  val = (val - m[f]) * s + be[f];
  out[(long)i * F + f] = fmaxf(val, 0.f);
}

// ---------------------------------------------------------------- attention
__global__ void attn_score(const int* __restrict__ src, const int* __restrict__ dst,
                           const float* __restrict__ q, const float* __restrict__ k,
                           float* __restrict__ score, int E) {
  int t = blockIdx.x * blockDim.x + threadIdx.x;
  if (t >= E * HEADS) return;
  int e = t >> 2, h = t & (HEADS - 1);
  int s = src[e], d = dst[e];
  const float* qd = q + (long)d * (HEADS * TH) + h * TH;
  const float* ks = k + (long)s * (HEADS * TH) + h * TH;
  float acc = 0.f;
#pragma unroll 4
  for (int j = 0; j < TH; j += 4) {
    float4 qa = *(const float4*)(qd + j);
    float4 ka = *(const float4*)(ks + j);
    acc += qa.x * ka.x + qa.y * ka.y + qa.z * ka.z + qa.w * ka.w;
  }
  score[t] = acc * 0.125f;  // 1/sqrt(64)
}

__device__ inline void atomicMaxF(float* addr, float val) {
  if (val >= 0.f) atomicMax((int*)addr, __float_as_int(val));
  else            atomicMin((unsigned int*)addr, __float_as_uint(val));
}

__global__ void attn_smax(const int* __restrict__ dst, const float* __restrict__ score,
                          float* __restrict__ smax, int E) {
  int t = blockIdx.x * blockDim.x + threadIdx.x;
  if (t >= E * HEADS) return;
  int e = t >> 2, h = t & (HEADS - 1);
  atomicMaxF(&smax[(long)dst[e] * HEADS + h], score[t]);
}

// ex = exp(score - smax[dst]) stored in-place; denom[dst,h] += ex
__global__ void attn_exp(const int* __restrict__ dst, float* __restrict__ score,
                         const float* __restrict__ smax, float* __restrict__ denom, int E) {
  int t = blockIdx.x * blockDim.x + threadIdx.x;
  if (t >= E * HEADS) return;
  int e = t >> 2, h = t & (HEADS - 1);
  int d = dst[e];
  float ex = expf(score[t] - smax[(long)d * HEADS + h]);
  score[t] = ex;
  atomicAdd(&denom[(long)d * HEADS + h], ex);
}

__global__ void attn_agg(const int* __restrict__ src, const int* __restrict__ dst,
                         const float* __restrict__ ex, const float* __restrict__ denom,
                         const float* __restrict__ v, float* __restrict__ agg, int E) {
  int e = blockIdx.x;
  int h = threadIdx.x >> 6;         // 0..3
  int f = threadIdx.x & (TH - 1);   // 0..63
  int s = src[e], d = dst[e];
  float alpha = ex[(long)e * HEADS + h] / (denom[(long)d * HEADS + h] + 1e-16f);
  atomicAdd(&agg[(long)d * (HEADS * TH) + h * TH + f],
            alpha * v[(long)s * (HEADS * TH) + h * TH + f]);
}

// head-mean + skip + BN + ReLU
__global__ void trans_fuse(const float* __restrict__ agg, const float* __restrict__ skip,
                           const float* __restrict__ g, const float* __restrict__ be,
                           const float* __restrict__ m, const float* __restrict__ v,
                           float* __restrict__ out) {
  int i = blockIdx.x;
  int f = threadIdx.x;  // 0..63
  const float* a = agg + (long)i * (HEADS * TH);
  float mean = (a[f] + a[TH + f] + a[2 * TH + f] + a[3 * TH + f]) * 0.25f;
  float val = mean + skip[(long)i * TH + f];
  float s = g[f] * rsqrtf(v[f] + BN_EPS);
  val = (val - m[f]) * s + be[f];
  out[(long)i * TH + f] = fmaxf(val, 0.f);
}

// ---------------------------------------------------------------- launcher
extern "C" void kernel_launch(void* const* d_in, const int* in_sizes, int n_in,
                              void* d_out, int out_size, void* d_ws, size_t ws_size,
                              hipStream_t stream) {
  const float* x   = (const float*)d_in[0];
  const int*  eidx = (const int*)d_in[1];
  const float* ew  = (const float*)d_in[2];
  const float* W1  = (const float*)d_in[3];
  const float* b1  = (const float*)d_in[4];
  const float* g1  = (const float*)d_in[5];
  const float* be1 = (const float*)d_in[6];
  const float* m1  = (const float*)d_in[7];
  const float* v1  = (const float*)d_in[8];
  const float* W2  = (const float*)d_in[9];
  const float* b2  = (const float*)d_in[10];
  const float* g2  = (const float*)d_in[11];
  const float* be2 = (const float*)d_in[12];
  const float* m2  = (const float*)d_in[13];
  const float* v2  = (const float*)d_in[14];
  const float* Wq  = (const float*)d_in[15];
  const float* bq  = (const float*)d_in[16];
  const float* Wk  = (const float*)d_in[17];
  const float* bk  = (const float*)d_in[18];
  const float* Wv  = (const float*)d_in[19];
  const float* bv  = (const float*)d_in[20];
  const float* Wsk = (const float*)d_in[21];
  const float* bsk = (const float*)d_in[22];
  const float* gt  = (const float*)d_in[23];
  const float* bet = (const float*)d_in[24];
  const float* mt  = (const float*)d_in[25];
  const float* vt  = (const float*)d_in[26];
  const float* Wfc = (const float*)d_in[27];
  const float* bfc = (const float*)d_in[28];

  const int N = in_sizes[0] / CIN;
  const int E = in_sizes[1] / 2;
  const int* src = eidx;
  const int* dst = eidx + E;

  // workspace carve-up (floats first, then bf16 packs; all 32B-aligned)
  float* W = (float*)d_ws;
  size_t o = 0;
  auto alloc = [&](size_t n) { float* p = W + o; o += n; return p; };
  float* deg   = alloc(N);
  float* dinv  = alloc(N);
  float* h1    = alloc((size_t)N * GH);
  float* agg1  = alloc((size_t)N * GH);
  float* h2    = alloc((size_t)N * GO);
  float* agg2  = alloc((size_t)N * GO);
  float* q     = alloc((size_t)N * HEADS * TH);
  float* kk    = alloc((size_t)N * HEADS * TH);
  float* vv    = alloc((size_t)N * HEADS * TH);
  float* skip  = alloc((size_t)N * TH);
  float* score = alloc((size_t)E * HEADS);
  float* smax  = alloc((size_t)N * HEADS);
  float* denom = alloc((size_t)N * HEADS);
  float* aggT  = alloc((size_t)N * HEADS * TH);
  float* zt    = alloc((size_t)N * TH);
  __bf16* bf_base = (__bf16*)(W + o);
  size_t ob = 0;
  auto balloc = [&](size_t n) { __bf16* p = bf_base + ob; ob += n; return p; };
  __bf16* W1p  = balloc((size_t)CIN * GH);
  __bf16* W2p  = balloc((size_t)GH * GO);
  __bf16* Wqp  = balloc((size_t)GO * HEADS * TH);
  __bf16* Wkp  = balloc((size_t)GO * HEADS * TH);
  __bf16* Wvp  = balloc((size_t)GO * HEADS * TH);
  __bf16* Wskp = balloc((size_t)GO * TH);
  __bf16* Wfcp = balloc((size_t)TH * COUT);
  (void)ws_size; (void)n_in; (void)out_size;

  const int TB = 256;
  auto cdiv = [](long a, long b) { return (int)((a + b - 1) / b); };
  const int Mtiles = cdiv(N, 16);
  dim3 gblk(32, 8);   // 8 waves/block, one N-tile each
  dim3 pblk(16, 32);  // pack kernel: j x lane

  // ---- pre-swizzle all weights into bf16 WMMA fragment order
  pack_B_bf16<CIN, GH><<<(GH / 16) * (CIN / 32), pblk, 0, stream>>>(W1, W1p);
  pack_B_bf16<GH, GO><<<(GO / 16) * (GH / 32), pblk, 0, stream>>>(W2, W2p);
  pack_B_bf16<GO, HEADS * TH><<<(HEADS * TH / 16) * (GO / 32), pblk, 0, stream>>>(Wq, Wqp);
  pack_B_bf16<GO, HEADS * TH><<<(HEADS * TH / 16) * (GO / 32), pblk, 0, stream>>>(Wk, Wkp);
  pack_B_bf16<GO, HEADS * TH><<<(HEADS * TH / 16) * (GO / 32), pblk, 0, stream>>>(Wv, Wvp);
  pack_B_bf16<GO, TH><<<(TH / 16) * (GO / 32), pblk, 0, stream>>>(Wsk, Wskp);
  pack_B_bf16<TH, COUT><<<(COUT / 16) * (TH / 32), pblk, 0, stream>>>(Wfc, Wfcp);

  // degree + symmetric norm (self loop weight 1 -> deg starts at 1)
  fill_kernel<<<cdiv(N, TB), TB, 0, stream>>>(deg, 1.0f, N);
  deg_accum_kernel<<<cdiv(E, TB), TB, 0, stream>>>(dst, ew, deg, E);
  dinv_kernel<<<cdiv(N, TB), TB, 0, stream>>>(deg, dinv, N);

  // ---- GCN layer 1: h1 = x @ W1 ; aggregate ; +b1, BN1, ReLU (in-place agg1)
  gemm_bf16_wmma<CIN, GH><<<dim3(Mtiles, 1), gblk, 0, stream>>>(x, W1p, nullptr, h1, N);
  fill_kernel<<<cdiv((long)N * GH, TB), TB, 0, stream>>>(agg1, 0.f, (long)N * GH);
  gcn_edge_agg<<<E, GH, 0, stream>>>(src, dst, ew, dinv, h1, agg1, GH);
  gcn_fuse<<<N, GH, 0, stream>>>(agg1, h1, dinv, b1, g1, be1, m1, v1, agg1, GH);

  // ---- GCN layer 2
  gemm_bf16_wmma<GH, GO><<<dim3(Mtiles, 1), gblk, 0, stream>>>(agg1, W2p, nullptr, h2, N);
  fill_kernel<<<cdiv((long)N * GO, TB), TB, 0, stream>>>(agg2, 0.f, (long)N * GO);
  gcn_edge_agg<<<E, GO, 0, stream>>>(src, dst, ew, dinv, h2, agg2, GO);
  gcn_fuse<<<N, GO, 0, stream>>>(agg2, h2, dinv, b2, g2, be2, m2, v2, agg2, GO);
  // agg2 is now z [N,64]

  // ---- TransformerConv projections
  gemm_bf16_wmma<GO, HEADS * TH><<<dim3(Mtiles, 2), gblk, 0, stream>>>(agg2, Wqp, bq, q, N);
  gemm_bf16_wmma<GO, HEADS * TH><<<dim3(Mtiles, 2), gblk, 0, stream>>>(agg2, Wkp, bk, kk, N);
  gemm_bf16_wmma<GO, HEADS * TH><<<dim3(Mtiles, 2), gblk, 0, stream>>>(agg2, Wvp, bv, vv, N);
  gemm_bf16_wmma<GO, TH><<<dim3(Mtiles, 1), gblk, 0, stream>>>(agg2, Wskp, bsk, skip, N);

  // ---- edge softmax + aggregation
  attn_score<<<cdiv((long)E * HEADS, TB), TB, 0, stream>>>(src, dst, q, kk, score, E);
  fill_kernel<<<cdiv((long)N * HEADS, TB), TB, 0, stream>>>(smax, -INFINITY, (long)N * HEADS);
  attn_smax<<<cdiv((long)E * HEADS, TB), TB, 0, stream>>>(dst, score, smax, E);
  fill_kernel<<<cdiv((long)N * HEADS, TB), TB, 0, stream>>>(denom, 0.f, (long)N * HEADS);
  attn_exp<<<cdiv((long)E * HEADS, TB), TB, 0, stream>>>(dst, score, smax, denom, E);
  fill_kernel<<<cdiv((long)N * HEADS * TH, TB), TB, 0, stream>>>(aggT, 0.f, (long)N * HEADS * TH);
  attn_agg<<<E, HEADS * TH, 0, stream>>>(src, dst, score, denom, vv, aggT, E);
  trans_fuse<<<N, TH, 0, stream>>>(aggT, skip, gt, bet, mt, vt, zt);

  // ---- final Linear -> d_out [N, 32]
  gemm_bf16_wmma<TH, COUT><<<dim3(Mtiles, 1), gblk, 0, stream>>>(zt, Wfcp, bfc, (float*)d_out, N);
}
